// SelfAttentionBlock_76055280877716
// MI455X (gfx1250) — matrix operand order repaired and checked by reading, not compiled, not run on hardware
//
#include <hip/hip_runtime.h>
#include <hip/hip_bf16.h>

#define BATCH 4
#define CCH   256
#define CQ    32
#define NTOK  4096
#define KT    128           // keys per flash-attention iteration

typedef __bf16 bf16;
typedef __attribute__((ext_vector_type(8)))  bf16  v8bf;
typedef __attribute__((ext_vector_type(16))) bf16  v16bf;
typedef __attribute__((ext_vector_type(8)))  float v8f;

static __device__ __forceinline__ unsigned pack2bf(float a, float b) {
  unsigned short ha = __builtin_bit_cast(unsigned short, (bf16)a);
  unsigned short hb = __builtin_bit_cast(unsigned short, (bf16)b);
  return (unsigned)ha | ((unsigned)hb << 16);
}

// A-operand fragment for v_wmma_f32_16x16x32_bf16 covering K=[0,32) of `row`.
// Lane (m, hi): elements 0..7 = row[hi*16 .. +7], 8..15 = row[hi*16+8 .. +7].
static __device__ __forceinline__ v16bf load_afrag(const bf16* row, int hi) {
  v8bf lo = *reinterpret_cast<const v8bf*>(row + hi * 16);
  v8bf hi8 = *reinterpret_cast<const v8bf*>(row + hi * 16 + 8);
  v16bf a;
#pragma unroll
  for (int i = 0; i < 8; ++i) { a[i] = lo[i]; a[i + 8] = hi8[i]; }
  return a;
}

static __device__ __forceinline__ v8f wmma_bf16(v16bf a, v16bf b, v8f c) {
  return __builtin_amdgcn_wmma_f32_16x16x32_bf16(false, a, false, b,
                                                 (short)0, c, false, false);
}

// ---------------- pack / convert kernels ----------------

__global__ void cvt_f32_bf16(const float* __restrict__ src, bf16* __restrict__ dst, int n) {
  int i = blockIdx.x * blockDim.x + threadIdx.x;
  if (i < n) dst[i] = (bf16)src[i];
}

// input [B,C,N] f32 -> xT [B,N,C] bf16 via 32x32 LDS tiles
__global__ void transpose_x(const float* __restrict__ x, bf16* __restrict__ xT) {
  __shared__ float tile[32][33];
  int b = blockIdx.z;
  int nb = blockIdx.x * 32, cb = blockIdx.y * 32;
  int tx = threadIdx.x, ty = threadIdx.y;
  const float* xb = x + ((size_t)b * CCH + cb) * NTOK + nb;
#pragma unroll
  for (int it = 0; it < 4; ++it) {
    int c = ty + it * 8;
    tile[c][tx] = xb[(size_t)c * NTOK + tx];
  }
  __syncthreads();
  bf16* xo = xT + ((size_t)b * NTOK + nb) * CCH + cb;
#pragma unroll
  for (int it = 0; it < 4; ++it) {
    int n = ty + it * 8;
    xo[(size_t)n * CCH + tx] = (bf16)tile[tx][n];
  }
}

// ---------------- projections (WMMA) ----------------

// fT/gT [B,N,32] = xT @ W^T + bias   (M=tokens, N=32, K=256)
__global__ void proj_fg(const bf16* __restrict__ xT,
                        const bf16* __restrict__ Wfb, const float* __restrict__ bf_,
                        const bf16* __restrict__ Wgb, const float* __restrict__ bg_,
                        bf16* __restrict__ fT, bf16* __restrict__ gT) {
  int sel = blockIdx.z;
  const bf16* W = sel ? Wgb : Wfb;
  const float* bias = sel ? bg_ : bf_;
  bf16* out = sel ? gT : fT;
  int b = blockIdx.y;
  int tb = blockIdx.x * 64;
  int w = threadIdx.x >> 5, lane = threadIdx.x & 31;
  int m = lane & 15, hi = lane >> 4;
  const bf16* arow = xT + ((size_t)b * NTOK + tb + w * 16 + m) * CCH;
  v8f acc[2] = {};
#pragma unroll
  for (int ks = 0; ks < 8; ++ks) {
    v16bf a = load_afrag(arow + ks * 32, hi);
#pragma unroll
    for (int ot = 0; ot < 2; ++ot) {
      int o = ot * 16 + m;
      v16bf bb = *reinterpret_cast<const v16bf*>(W + (size_t)o * CCH + ks * 32 + hi * 16);
      acc[ot] = wmma_bf16(a, bb, acc[ot]);
    }
  }
#pragma unroll
  for (int ot = 0; ot < 2; ++ot) {
    int o = ot * 16 + m;
    float bv = bias[o];
#pragma unroll
    for (int r = 0; r < 8; ++r) {
      int tok = tb + w * 16 + r + 8 * hi;
      out[((size_t)b * NTOK + tok) * CQ + o] = (bf16)(acc[ot][r] + bv);
    }
  }
}

// h [B,C,N] = Wh @ x + bh   (A = Wh rows, B^T = xT rows)
__global__ void proj_h(const bf16* __restrict__ xT, const bf16* __restrict__ Whb,
                       const float* __restrict__ bh_, bf16* __restrict__ hB) {
  int b = blockIdx.z;
  int cb = blockIdx.y * 64;
  int nb = blockIdx.x * 128;
  int w = threadIdx.x >> 5, lane = threadIdx.x & 31;
  int m = lane & 15, hi = lane >> 4;
  int mt = w >> 1;
  int ntb = (w & 1) * 4;
  const bf16* arow = Whb + (size_t)(cb + mt * 16 + m) * CCH;
  v8f acc[4] = {};
#pragma unroll
  for (int ks = 0; ks < 8; ++ks) {
    v16bf a = load_afrag(arow + ks * 32, hi);
#pragma unroll
    for (int j = 0; j < 4; ++j) {
      int n = nb + (ntb + j) * 16 + m;
      v16bf bb = *reinterpret_cast<const v16bf*>(
          xT + ((size_t)b * NTOK + n) * CCH + ks * 32 + hi * 16);
      acc[j] = wmma_bf16(a, bb, acc[j]);
    }
  }
#pragma unroll
  for (int j = 0; j < 4; ++j) {
#pragma unroll
    for (int r = 0; r < 8; ++r) {
      int c = cb + mt * 16 + r + 8 * hi;
      int n = nb + (ntb + j) * 16 + m;
      hB[((size_t)b * CCH + c) * NTOK + n] = (bf16)(acc[j][r] + bh_[c]);
    }
  }
}

// ---------------- fused flash attention ----------------

__global__ void __launch_bounds__(256)
attn(const bf16* __restrict__ fT, const bf16* __restrict__ gT,
     const bf16* __restrict__ hB, const float* __restrict__ inp,
     const float* __restrict__ gamma, float* __restrict__ outp) {
  __shared__ float S[64][KT + 4];      // S tile / reused as O staging
  __shared__ bf16  P[64][KT];          // probabilities (A-operand source)
  __shared__ float red[4][64];
  __shared__ float row_m[64], row_l[64], row_s[64];

  int b = blockIdx.y;
  int qb = blockIdx.x * 64;
  int t = threadIdx.x;
  int w = t >> 5, lane = t & 31;
  int m = lane & 15, hi = lane >> 4;
  int qt = w >> 1;        // query tile (0..3) of this wave
  int half = w & 1;       // key-tile quad for S / channel half for PV

  if (t < 64) { row_m[t] = -__builtin_inff(); row_l[t] = 0.f; }
  __syncthreads();

  // Q fragment: dk=32 fits one A-frag
  v16bf aq = load_afrag(fT + ((size_t)b * NTOK + qb + qt * 16 + m) * CQ, hi);

  v8f o[8];
#pragma unroll
  for (int j = 0; j < 8; ++j) o[j] = {};

  const int srow = t & 63, sq = t >> 6;   // softmax row / column quarter (32 cols)

  for (int kb = 0; kb < NTOK; kb += KT) {
    // ---- S = Q @ K^T, 4 tiles per wave ----
#pragma unroll
    for (int j = 0; j < 4; ++j) {
      int kt = half * 4 + j;
      v16bf bk = *reinterpret_cast<const v16bf*>(
          gT + ((size_t)b * NTOK + kb + kt * 16 + m) * CQ + hi * 16);
      v8f s = wmma_bf16(aq, bk, v8f{});
#pragma unroll
      for (int r = 0; r < 8; ++r)
        S[qt * 16 + r + 8 * hi][kt * 16 + m] = s[r];
    }
    __syncthreads();

    // ---- online softmax pass 1: row max over this key tile ----
    const float4* Sv = reinterpret_cast<const float4*>(&S[srow][sq * 32]);
    float pm = -__builtin_inff();
#pragma unroll
    for (int i = 0; i < 8; ++i) {
      float4 v = Sv[i];
      pm = fmaxf(pm, fmaxf(fmaxf(v.x, v.y), fmaxf(v.z, v.w)));
    }
    red[sq][srow] = pm;
    __syncthreads();
    if (t < 64) {
      float tmax = fmaxf(fmaxf(red[0][t], red[1][t]), fmaxf(red[2][t], red[3][t]));
      float oldm = row_m[t];
      float nm = fmaxf(oldm, tmax);
      float sc = __expf(oldm - nm);
      row_s[t] = sc; row_m[t] = nm; row_l[t] *= sc;
    }
    __syncthreads();

    // ---- pass 2: P = exp(S - m) packed bf16x2, partial sums ----
    {
      float nm = row_m[srow];
      float ps = 0.f;
      unsigned* Pu = reinterpret_cast<unsigned*>(&P[srow][0]);
#pragma unroll
      for (int i = 0; i < 8; ++i) {
        float4 v = Sv[i];
        float p0 = __expf(v.x - nm), p1 = __expf(v.y - nm);
        float p2 = __expf(v.z - nm), p3 = __expf(v.w - nm);
        ps += (p0 + p1) + (p2 + p3);
        Pu[sq * 16 + i * 2]     = pack2bf(p0, p1);
        Pu[sq * 16 + i * 2 + 1] = pack2bf(p2, p3);
      }
      red[sq][srow] = ps;
    }
    __syncthreads();
    if (t < 64) row_l[t] += red[0][t] + red[1][t] + red[2][t] + red[3][t];

    // ---- rescale running O by exp(m_old - m_new) ----
#pragma unroll
    for (int r = 0; r < 8; ++r) {
      float sc = row_s[qt * 16 + r + 8 * hi];
#pragma unroll
      for (int j = 0; j < 8; ++j) o[j][r] *= sc;
    }

    // ---- O += P @ V (V B-frags from channel-major h in L2) ----
#pragma unroll
    for (int kh = 0; kh < 4; ++kh) {
      v16bf ap = load_afrag(&P[qt * 16 + m][kh * 32], hi);
#pragma unroll
      for (int j = 0; j < 8; ++j) {
        int c = (half * 8 + j) * 16 + m;
        v16bf bv = *reinterpret_cast<const v16bf*>(
            hB + ((size_t)b * CCH + c) * NTOK + kb + kh * 32 + hi * 16);
        o[j] = wmma_bf16(ap, bv, o[j]);
      }
    }
    __syncthreads();
  }

  // ---- normalize, then gamma*O + input, coalesced via LDS transpose ----
#pragma unroll
  for (int r = 0; r < 8; ++r) {
    float inv = 1.f / row_l[qt * 16 + r + 8 * hi];
#pragma unroll
    for (int j = 0; j < 8; ++j) o[j][r] *= inv;
  }

  float gm = gamma[0];
  for (int g = 0; g < 4; ++g) {        // 64 channels per group
    __syncthreads();
    if (half == (g >> 1)) {
#pragma unroll
      for (int jj = 0; jj < 4; ++jj) {
        int fi = (g & 1) * 4 + jj;     // local frag index within wave
#pragma unroll
        for (int r = 0; r < 8; ++r) {
          int q = qt * 16 + r + 8 * hi;
          S[jj * 16 + m][q] = o[fi][r];
        }
      }
    }
    __syncthreads();
#pragma unroll
    for (int it = 0; it < 16; ++it) {
      int idx = it * 256 + t;
      int chl = idx >> 6, q = idx & 63;
      size_t off = ((size_t)b * CCH + g * 64 + chl) * (size_t)NTOK + qb + q;
      outp[off] = gm * S[chl][q] + inp[off];
    }
  }
}

// ---------------- host launch ----------------

extern "C" void kernel_launch(void* const* d_in, const int* in_sizes, int n_in,
                              void* d_out, int out_size, void* d_ws, size_t ws_size,
                              hipStream_t stream) {
  const float* input = (const float*)d_in[0];
  const float* Wf = (const float*)d_in[1];
  const float* bf_ = (const float*)d_in[2];
  const float* Wg = (const float*)d_in[3];
  const float* bg_ = (const float*)d_in[4];
  const float* Wh = (const float*)d_in[5];
  const float* bh_ = (const float*)d_in[6];
  const float* gamma = (const float*)d_in[7];
  float* out = (float*)d_out;

  char* ws = (char*)d_ws;
  size_t off = 0;
  auto alloc = [&](size_t bytes) -> void* {
    off = (off + 255) & ~(size_t)255;
    void* p = ws + off;
    off += bytes;
    return p;
  };
  bf16* xT  = (bf16*)alloc((size_t)BATCH * NTOK * CCH * 2);
  bf16* Wfb = (bf16*)alloc((size_t)CQ * CCH * 2);
  bf16* Wgb = (bf16*)alloc((size_t)CQ * CCH * 2);
  bf16* Whb = (bf16*)alloc((size_t)CCH * CCH * 2);
  bf16* fTd = (bf16*)alloc((size_t)BATCH * NTOK * CQ * 2);
  bf16* gTd = (bf16*)alloc((size_t)BATCH * NTOK * CQ * 2);
  bf16* hBd = (bf16*)alloc((size_t)BATCH * CCH * NTOK * 2);
  (void)ws_size; (void)in_sizes; (void)n_in; (void)out_size;

  cvt_f32_bf16<<<dim3((CQ * CCH + 255) / 256), 256, 0, stream>>>(Wf, Wfb, CQ * CCH);
  cvt_f32_bf16<<<dim3((CQ * CCH + 255) / 256), 256, 0, stream>>>(Wg, Wgb, CQ * CCH);
  cvt_f32_bf16<<<dim3((CCH * CCH + 255) / 256), 256, 0, stream>>>(Wh, Whb, CCH * CCH);
  transpose_x<<<dim3(NTOK / 32, CCH / 32, BATCH), dim3(32, 8), 0, stream>>>(input, xT);
  proj_fg<<<dim3(NTOK / 64, BATCH, 2), 128, 0, stream>>>(xT, Wfb, bf_, Wgb, bg_, fTd, gTd);
  proj_h<<<dim3(NTOK / 128, CCH / 64, BATCH), 256, 0, stream>>>(xT, Whb, bh_, hBd);
  attn<<<dim3(NTOK / 64, BATCH), 256, 0, stream>>>(fTd, gTd, hBd, input, gamma, out);
}
